// MultiObserverGNN_45672682225672
// MI455X (gfx1250) — matrix operand order repaired and checked by reading, not compiled, not run on hardware
//
#include <hip/hip_runtime.h>

// ---------------- problem constants ----------------
#define NN 50000
#define EE 800000
#define HH 128      // F = H = O = 128
#define KK 4

// ---------------- ws layout (float units) ----------------
#define HW_OFF    ((size_t)0)                       // hw: N*H f32
#define AGG_OFF   ((size_t)NN * HH)                 // agg: K*N*H f32
#define DIST_OFF  (AGG_OFF + (size_t)KK * NN * HH)  // dist: E
#define DSUM_OFF  (DIST_OFF + (size_t)EE)           // dsum: E
#define SUM_OFF   (DSUM_OFF + (size_t)EE)           // sum: K*H
#define SQ_OFF    (SUM_OFF + (size_t)KK * HH)       // sumsq: K*H
#define SC_OFF    (SQ_OFF + (size_t)KK * HH)        // bn scale: K*H
#define SH_OFF    (SC_OFF + (size_t)KK * HH)        // bn shift: K*H
#define MAX_OFF   (SH_OFF + (size_t)KK * HH)        // maxdist bits: 1
// bf16 staging (2 bf16 per float slot)
#define XBF_OFF   (MAX_OFF + 1)                            // N*H bf16
#define WFT_OFF   (XBF_OFF  + (size_t)NN * HH / 2)         // H*H bf16
#define WCV_OFF   (WFT_OFF  + (size_t)HH * HH / 2)         // H*H bf16
#define W1B_OFF   (WCV_OFF  + (size_t)HH * HH / 2)         // 256*512 bf16
#define W2B_OFF   (W1B_OFF  + (size_t)256 * 512 / 2)       // 128*256 bf16

#define AGGN  ((size_t)KK * NN * HH)
#define TAILN ((size_t)(4 * KK * HH + 1))           // sum,sumsq,scale,shift,maxbits

typedef __attribute__((ext_vector_type(16))) __bf16 v16bf;
typedef __attribute__((ext_vector_type(8)))  __bf16 v8bf;
typedef __attribute__((ext_vector_type(8)))  float  v8f;

__device__ __forceinline__ v8f wmma_bf16(v16bf a, v16bf b, v8f c) {
  return __builtin_amdgcn_wmma_f32_16x16x32_bf16(false, a, false, b, (short)0, c,
                                                 false, false);
}

// A-fragment (16-bit A 16x32, ISA 7.12.2): lane m = lane&15.
// half==0: elems 0..7 -> K=kb+0..7,  elems 8..15 -> K=kb+16..23
// half==1: elems 0..7 -> K=kb+8..15, elems 8..15 -> K=kb+24..31
// row pointers are 16B-aligned at every (kb,hf) offset -> two v8bf loads.
__device__ __forceinline__ v16bf afrag_bf(const __bf16* row, int kb, int hf) {
  v8bf lo = *(const v8bf*)(row + kb + hf * 8);
  v8bf hi = *(const v8bf*)(row + kb + 16 + hf * 8);
  return __builtin_shufflevector(lo, hi, 0, 1, 2, 3, 4, 5, 6, 7,
                                 8, 9, 10, 11, 12, 13, 14, 15);
}
// B-fragment (16-bit B 32x16): lane col n = lane&15, same K order as A-fragment:
// elems 0..7 -> K=kb+8*hf+0..7, elems 8..15 -> K=kb+16+8*hf+0..7 (16B loads).
__device__ __forceinline__ v16bf bfrag_bf(const __bf16* wrow, int kb, int hf) {
  v8bf lo = *(const v8bf*)(wrow + kb + hf * 8);
  v8bf hi = *(const v8bf*)(wrow + kb + 16 + hf * 8);
  return __builtin_shufflevector(lo, hi, 0, 1, 2, 3, 4, 5, 6, 7,
                                 8, 9, 10, 11, 12, 13, 14, 15);
}

// ============ one-time f32 -> bf16 staging ============
__global__ void k_cvt(const float* __restrict__ src, __bf16* __restrict__ dst,
                      int n) {
  for (int i = blockIdx.x * blockDim.x + threadIdx.x; i < n;
       i += gridDim.x * blockDim.x)
    dst[i] = (__bf16)src[i];
}

// ============ kernel 1: hw = (relu(x@Wft^T + bft)) @ Wcv^T + bcv ============
// one wave per 16-node tile; h kept as bf16 in LDS between the two WMMA passes
__global__ __launch_bounds__(32) void k_feat_conv(const __bf16* __restrict__ xbf,
    const __bf16* __restrict__ Wft, const float* __restrict__ bft,
    const __bf16* __restrict__ Wcv, const float* __restrict__ bcv,
    float* __restrict__ hw) {
  __shared__ __bf16 hbuf[16 * HH];
  const int lane = threadIdx.x & 31;
  const int hf = lane >> 4;
  const int m  = lane & 15;
  const int node0 = blockIdx.x * 16;

  // ---- pass 1: h = relu(x @ Wft^T + bft) -> LDS (bf16)
  const __bf16* xrow = xbf + (size_t)(node0 + m) * HH;
  v16bf af[4];
#pragma unroll
  for (int kb = 0; kb < 4; ++kb) af[kb] = afrag_bf(xrow, kb * 32, hf);

#pragma unroll
  for (int nt = 0; nt < 8; ++nt) {
    const int col = nt * 16 + m;                    // n == lane&15
    const __bf16* wrow = Wft + (size_t)col * HH;
    v8f acc = {};
#pragma unroll
    for (int kb = 0; kb < 4; ++kb)
      acc = wmma_bf16(af[kb], bfrag_bf(wrow, kb * 32, hf), acc);
    const float bias = bft[col];
#pragma unroll
    for (int g = 0; g < 8; ++g) {
      float v = acc[g] + bias;
      v = v > 0.f ? v : 0.f;
      hbuf[(g + hf * 8) * HH + col] = (__bf16)v;
    }
  }
  __syncthreads();

  // ---- pass 2: hw = h @ Wcv^T + bcv -> global (f32)
  const __bf16* hrow = hbuf + m * HH;
  v16bf ah[4];
#pragma unroll
  for (int kb = 0; kb < 4; ++kb) ah[kb] = afrag_bf(hrow, kb * 32, hf);

#pragma unroll
  for (int nt = 0; nt < 8; ++nt) {
    const int col = nt * 16 + m;
    const __bf16* wrow = Wcv + (size_t)col * HH;
    v8f acc = {};
#pragma unroll
    for (int kb = 0; kb < 4; ++kb)
      acc = wmma_bf16(ah[kb], bfrag_bf(wrow, kb * 32, hf), acc);
    const float bias = bcv[col];
#pragma unroll
    for (int g = 0; g < 8; ++g)
      hw[(size_t)(node0 + g + hf * 8) * HH + col] = acc[g] + bias;
  }
}

// ============ workspace zeroing ============
__global__ void k_zero(float* __restrict__ agg, float* __restrict__ tail) {
  const size_t total = AGGN + TAILN;
  for (size_t i = (size_t)blockIdx.x * blockDim.x + threadIdx.x; i < total;
       i += (size_t)gridDim.x * blockDim.x) {
    if (i < AGGN) agg[i] = 0.f;
    else          tail[i - AGGN] = 0.f;
  }
}

// ============ per-edge geometry + global max(dist) ============
__global__ __launch_bounds__(256) void k_edge_pre(const int* __restrict__ ei,
    const float* __restrict__ pos, float* __restrict__ dist,
    float* __restrict__ dsum, unsigned* __restrict__ maxbits) {
  __shared__ float red[256];
  const int e = blockIdx.x * 256 + threadIdx.x;    // E % 256 == 0
  const int s = ei[e], t = ei[EE + e];
  const float dx = pos[s * 3 + 0] - pos[t * 3 + 0];
  const float dy = pos[s * 3 + 1] - pos[t * 3 + 1];
  const float dz = pos[s * 3 + 2] - pos[t * 3 + 2];
  const float d = sqrtf(dx * dx + dy * dy + dz * dz);
  dist[e] = d;
  dsum[e] = (dx + dy + dz) / (d + 1e-8f);
  red[threadIdx.x] = d;
  __syncthreads();
#pragma unroll
  for (int w = 128; w > 0; w >>= 1) {
    if (threadIdx.x < w)
      red[threadIdx.x] = fmaxf(red[threadIdx.x], red[threadIdx.x + w]);
    __syncthreads();
  }
  if (threadIdx.x == 0) atomicMax(maxbits, __float_as_uint(red[0]));
}

// ============ edge gather + K-observer scatter-add (L2-resident agg) ========
__global__ __launch_bounds__(256) void k_edge_scatter(const int* __restrict__ ei,
    const float* __restrict__ dist, const float* __restrict__ dsum,
    const unsigned* __restrict__ maxbits, const float* __restrict__ vfac,
    const float* __restrict__ vmax, const float* __restrict__ hw,
    float* __restrict__ agg) {
  const int lane = threadIdx.x & 31;
  const int e = blockIdx.x * 8 + (threadIdx.x >> 5);  // E/8 blocks exactly
  const float maxd = __uint_as_float(*maxbits);
  const float r  = dist[e] / maxd;
  const float dS = dsum[e];
  const float vf = vfac[0];
  float ab[KK];
#pragma unroll
  for (int i = 0; i < KK; ++i) {
    const float v = vf * fminf(r, vmax[i]);          // r >= 0 already
    // 1/(gamma*(1+v*dsum)) with gamma = 1/sqrt(1-v^2+1e-8)
    ab[i] = sqrtf(1.f - v * v + 1e-8f) / (1.f + v * dS);
  }
  const int s = ei[e], t = ei[EE + e];
  const float4 msg = ((const float4*)(hw + (size_t)s * HH))[lane];
  const size_t base = (size_t)t * HH + lane * 4;
#pragma unroll
  for (int i = 0; i < KK; ++i) {
    float* ap = agg + (size_t)i * NN * HH + base;
    atomicAdd(ap + 0, msg.x * ab[i]);
    atomicAdd(ap + 1, msg.y * ab[i]);
    atomicAdd(ap + 2, msg.z * ab[i]);
    atomicAdd(ap + 3, msg.w * ab[i]);
  }
}

// ============ BN statistics: per-(observer,feature) sum / sumsq over N ======
__global__ __launch_bounds__(128) void k_bn_stats(const float* __restrict__ agg,
    float* __restrict__ sum, float* __restrict__ sumsq) {
  const int f = threadIdx.x;
  const int i = blockIdx.y;
  const int n0 = blockIdx.x * 128;
  const int n1 = min(n0 + 128, NN);
  float s = 0.f, s2 = 0.f;
  for (int n = n0; n < n1; ++n) {                    // rows are 512B-contiguous
    const float v = agg[((size_t)i * NN + n) * HH + f];
    s += v; s2 += v * v;
  }
  atomicAdd(&sum[i * HH + f], s);
  atomicAdd(&sumsq[i * HH + f], s2);
}

__global__ void k_bn_finalize(const float* __restrict__ sum,
    const float* __restrict__ sumsq, const float* __restrict__ g,
    const float* __restrict__ b, float* __restrict__ scale,
    float* __restrict__ shift) {
  const int idx = blockIdx.x * blockDim.x + threadIdx.x;
  if (idx >= KK * HH) return;
  const float inv = 1.f / (float)NN;
  const float mu  = sum[idx] * inv;
  const float var = sumsq[idx] * inv - mu * mu;      // biased, as reference
  const float sc  = g[idx] * rsqrtf(var + 1e-5f);
  scale[idx] = sc;
  shift[idx] = b[idx] - mu * sc;
}

// ============ fused BN-norm/ReLU/attn + W1 GEMM + LN + ReLU + W2 GEMM =======
// one wave per 16-node tile; zA (bf16 16x512) / zB (f32 16x256) in LDS
__global__ __launch_bounds__(32) void k_mlp(const float* __restrict__ agg,
    const float* __restrict__ scale, const float* __restrict__ shift,
    const float* __restrict__ att, const __bf16* __restrict__ W1,
    const float* __restrict__ b1, const float* __restrict__ lng,
    const float* __restrict__ lnb, const __bf16* __restrict__ W2,
    const float* __restrict__ b2, float* __restrict__ out) {
  __shared__ __bf16 zA[16 * 512];
  __shared__ float  zB[16 * 256];
  const int lane = threadIdx.x & 31;
  const int hf = lane >> 4;
  const int m  = lane & 15;
  const int node0 = blockIdx.x * 16;

  // ---- stage 1: z = relu(bn(agg))*att -> zA (bf16), coalesced 128B/row
  for (int it = 0; it < 256; ++it) {
    const int idx = it * 32 + lane;                  // 0..8191
    const int row = idx >> 9;
    const int col = idx & 511;
    const int i = col >> 7, f = col & 127;
    float v = agg[((size_t)i * NN + node0 + row) * HH + f];
    v = v * scale[i * HH + f] + shift[i * HH + f];
    v = v > 0.f ? v : 0.f;
    v *= att[i * HH + f];
    zA[idx] = (__bf16)v;
  }
  __syncthreads();

  // ---- stage 2: z1 = z @ W1^T + b1  (K=512, 16 n-tiles) -> zB (f32)
  const __bf16* arow = zA + m * 512;
#pragma unroll
  for (int nt = 0; nt < 16; ++nt) {
    const int col = nt * 16 + m;
    const __bf16* wrow = W1 + (size_t)col * 512;
    v8f acc = {};
#pragma unroll
    for (int kb = 0; kb < 16; ++kb)
      acc = wmma_bf16(afrag_bf(arow, kb * 32, hf), bfrag_bf(wrow, kb * 32, hf),
                      acc);
    const float bias = b1[col];
#pragma unroll
    for (int g = 0; g < 8; ++g)
      zB[(g + hf * 8) * 256 + col] = acc[g] + bias;
  }
  __syncthreads();

  // ---- stage 3: row LayerNorm + ReLU -> z2 (bf16, reuse zA)
  __bf16* z2 = zA;
  for (int r = 0; r < 16; ++r) {
    float s = 0.f, s2 = 0.f;
#pragma unroll
    for (int c = lane; c < 256; c += 32) {
      const float v = zB[r * 256 + c];
      s += v; s2 += v * v;
    }
#pragma unroll
    for (int off = 16; off > 0; off >>= 1) {
      s  += __shfl_xor(s,  off, 32);
      s2 += __shfl_xor(s2, off, 32);
    }
    const float mu  = s * (1.f / 256.f);
    const float var = s2 * (1.f / 256.f) - mu * mu;
    const float rs  = rsqrtf(var + 1e-5f);
#pragma unroll
    for (int c = lane; c < 256; c += 32) {
      float v = (zB[r * 256 + c] - mu) * rs * lng[c] + lnb[c];
      v = v > 0.f ? v : 0.f;
      z2[r * 256 + c] = (__bf16)v;
    }
  }
  __syncthreads();

  // ---- stage 4: out = z2 @ W2^T + b2  (K=256, 8 n-tiles)
  const __bf16* a2 = z2 + m * 256;
#pragma unroll
  for (int nt = 0; nt < 8; ++nt) {
    const int col = nt * 16 + m;
    const __bf16* wrow = W2 + (size_t)col * 256;
    v8f acc = {};
#pragma unroll
    for (int kb = 0; kb < 8; ++kb)
      acc = wmma_bf16(afrag_bf(a2, kb * 32, hf), bfrag_bf(wrow, kb * 32, hf),
                      acc);
    const float bias = b2[col];
#pragma unroll
    for (int g = 0; g < 8; ++g)
      out[(size_t)(node0 + g + hf * 8) * HH + col] = acc[g] + bias;
  }
}

// ---------------------------------------------------------------------------
extern "C" void kernel_launch(void* const* d_in, const int* in_sizes, int n_in,
                              void* d_out, int out_size, void* d_ws, size_t ws_size,
                              hipStream_t stream) {
  const float* x     = (const float*)d_in[0];
  const int*   ei    = (const int*)  d_in[1];
  const float* pos   = (const float*)d_in[2];
  const float* Wft   = (const float*)d_in[3];
  const float* bft   = (const float*)d_in[4];
  const float* Wcv   = (const float*)d_in[5];
  const float* bcv   = (const float*)d_in[6];
  const float* vfac  = (const float*)d_in[7];
  const float* vmax  = (const float*)d_in[8];
  const float* bng   = (const float*)d_in[9];
  const float* bnb   = (const float*)d_in[10];
  const float* att   = (const float*)d_in[11];
  const float* W1    = (const float*)d_in[12];
  const float* b1    = (const float*)d_in[13];
  const float* lng   = (const float*)d_in[14];
  const float* lnb   = (const float*)d_in[15];
  const float* W2    = (const float*)d_in[16];
  const float* b2    = (const float*)d_in[17];
  float* out = (float*)d_out;

  float* ws   = (float*)d_ws;
  float* hw   = ws + HW_OFF;
  float* agg  = ws + AGG_OFF;
  float* dist = ws + DIST_OFF;
  float* dsum = ws + DSUM_OFF;
  float* sum  = ws + SUM_OFF;
  float* sq   = ws + SQ_OFF;
  float* sc   = ws + SC_OFF;
  float* sh   = ws + SH_OFF;
  unsigned* maxbits = (unsigned*)(ws + MAX_OFF);
  __bf16* xbf   = (__bf16*)(ws + XBF_OFF);
  __bf16* Wftb  = (__bf16*)(ws + WFT_OFF);
  __bf16* Wcvb  = (__bf16*)(ws + WCV_OFF);
  __bf16* W1b   = (__bf16*)(ws + W1B_OFF);
  __bf16* W2b   = (__bf16*)(ws + W2B_OFF);

  // 0) zero agg + reduction tail (maxbits=0 == float 0.0)
  k_zero<<<16384, 256, 0, stream>>>(agg, sum);

  // 0b) one-time bf16 staging of activations-in and all weights
  k_cvt<<<4096, 256, 0, stream>>>(x,   xbf,  NN * HH);
  k_cvt<<<64,   256, 0, stream>>>(Wft, Wftb, HH * HH);
  k_cvt<<<64,   256, 0, stream>>>(Wcv, Wcvb, HH * HH);
  k_cvt<<<512,  256, 0, stream>>>(W1,  W1b,  256 * 512);
  k_cvt<<<128,  256, 0, stream>>>(W2,  W2b,  128 * 256);

  // 1) fused node GEMMs: hw = relu(x@Wft^T+bft)@Wcv^T+bcv
  k_feat_conv<<<NN / 16, 32, 0, stream>>>(xbf, Wftb, bft, Wcvb, bcv, hw);

  // 2) edge geometry + max distance
  k_edge_pre<<<EE / 256, 256, 0, stream>>>(ei, pos, dist, dsum, maxbits);

  // 3) gather/scatter with K observer factors (wave per edge)
  k_edge_scatter<<<EE / 8, 256, 0, stream>>>(ei, dist, dsum, maxbits, vfac,
                                             vmax, hw, agg);

  // 4) BN stats + fold into scale/shift
  k_bn_stats<<<dim3((NN + 127) / 128, KK), 128, 0, stream>>>(agg, sum, sq);
  k_bn_finalize<<<(KK * HH + 255) / 256, 256, 0, stream>>>(sum, sq, bng, bnb, sc, sh);

  // 5) fused BN-norm/attn + MLP (W1 -> LN -> ReLU -> W2)
  k_mlp<<<NN / 16, 32, 0, stream>>>(agg, sc, sh, att, W1b, b1, lng, lnb, W2b, b2, out);
}
